// TemporalUnitRolloutV341_47047071760682
// MI455X (gfx1250) — matrix operand order repaired and checked by reading, not compiled, not run on hardware
//
#include <hip/hip_runtime.h>

// B=8, U=256, D=512, H=32  ->  BU=2048 rows, 3D=1536 gate width.
#define HZN 32

typedef __attribute__((ext_vector_type(16))) __bf16 v16bf;
typedef __attribute__((ext_vector_type(8)))  float  v8f;

__device__ inline __bf16 f2bf(float f) {
  union { float f; unsigned u; } v; v.f = f;
  unsigned r = v.u + 0x7FFFu + ((v.u >> 16) & 1u);   // round-to-nearest-even
  unsigned short h = (unsigned short)(r >> 16);
  return __builtin_bit_cast(__bf16, h);
}

__device__ inline v8f wmma_bf16(v16bf a, v16bf b, v8f c) {
  // (neg_a, A, neg_b, B, c_mod, C, reuse_a, reuse_b)
  return __builtin_amdgcn_wmma_f32_16x16x32_bf16(false, a, false, b, (short)0, c,
                                                 false, false);
}

// 16-bit A-matrix 16x32 layout (ISA 7.12.2): lane lo = row M, VGPR i holds K
// pairs; lanes 0-15: K = {0..7,16..23}, lanes 16-31: +8.  Per lane this is two
// contiguous 16B spans -> compiler emits 2x ds_load_b128 (verified in asm).
__device__ inline v16bf load_a_frag(const __bf16* tile, int ldk, int kbase,
                                    int lo, int hi) {
  v16bf a;
  const __bf16* rp = tile + lo * ldk + kbase + 8 * hi;
#pragma unroll
  for (int i = 0; i < 8; ++i) {
    int koff = (i < 4) ? (2 * i) : (2 * i + 8);
    a[2 * i]     = rp[koff];
    a[2 * i + 1] = rp[koff + 1];
  }
  return a;
}

// B fragments pre-packed so each lane's 16 bf16 are contiguous (32B aligned)
// -> 2x global_load_b128 per fragment.
__device__ inline v16bf load_b_frag(const __bf16* pk, int nt, int kc, int nk,
                                    int lane) {
  return *(const v16bf*)(pk + ((((size_t)nt * nk + kc) * 32) + lane) * 16);
}

__device__ inline float sigm(float x) { return 1.f / (1.f + __expf(-x)); }
__device__ inline float tanh_f(float x) {
  x = fminf(fmaxf(x, -15.f), 15.f);
  float e = __expf(2.f * x);
  return (e - 1.f) / (e + 1.f);
}

// ---- Pack weight [N][K] (f32) into bf16 WMMA-B fragment layout ----
// frag (nt, kc): element (lane, p) = w[nt*16 + (lane&15)][kc*32 + (p&7) + 8*(lane>>4) + 16*(p>>3)]
__global__ __launch_bounds__(256) void pack_kernel(const float* __restrict__ w,
                                                   __bf16* __restrict__ dst,
                                                   int N, int K) {
  int nk = K / 32;
  size_t total = (size_t)N * K;
  for (size_t e = (size_t)blockIdx.x * blockDim.x + threadIdx.x; e < total;
       e += (size_t)gridDim.x * blockDim.x) {
    int p    = (int)(e & 15);
    int lane = (int)((e >> 4) & 31);
    size_t rem = e >> 9;
    int kc = (int)(rem % nk);
    int nt = (int)(rem / nk);
    int n  = nt * 16 + (lane & 15);
    int kk = kc * 32 + (p & 7) + 8 * (lane >> 4) + 16 * (p >> 3);
    dst[e] = f2bf(w[(size_t)n * K + kk]);
  }
}

// ---- tprojb[t][j] = b_ih[j] + sum_k te_w[min(t,31)][k] * w_ih[j][k] ----
__global__ __launch_bounds__(256) void tproj_kernel(const float* __restrict__ te_w,
                                                    const float* __restrict__ w_ih,
                                                    const float* __restrict__ b_ih,
                                                    float* __restrict__ tprojb) {
  int e = blockIdx.x * blockDim.x + threadIdx.x;
  if (e >= HZN * 1536) return;
  int tt = e / 1536, col = e % 1536;
  int idx = tt < 32 ? tt : 31;
  float s = b_ih[col];
  for (int k = 0; k < 512; ++k)
    s += te_w[idx * 512 + k] * w_ih[(size_t)col * 512 + k];
  tprojb[e] = s;
}

// ---- gi0[2048,1536] = sem_bf16 @ wih_pk (WMMA) ----
__global__ __launch_bounds__(256) void gi0_kernel(const float* __restrict__ z_sem,
                                                  const __bf16* __restrict__ wih_pk,
                                                  float* __restrict__ gi0) {
  __shared__ __bf16 sbf[16 * 512];
  const int tid = threadIdx.x;
  const int wave = tid >> 5, lane = tid & 31;
  const int lo = lane & 15, hi = lane >> 4;
  const int orow = tid & 15, oc0 = (tid >> 4) * 32;
  const int m0 = blockIdx.x * 16;
#pragma unroll
  for (int j = 0; j < 32; ++j)
    sbf[orow * 512 + oc0 + j] = f2bf(z_sem[(size_t)(m0 + orow) * 512 + oc0 + j]);
  __syncthreads();
  const v8f z8 = {0.f, 0.f, 0.f, 0.f, 0.f, 0.f, 0.f, 0.f};
  v8f acc[12];
#pragma unroll
  for (int q = 0; q < 12; ++q) acc[q] = z8;
  v16bf a = load_a_frag(sbf, 512, 0, lo, hi);
  for (int k = 0; k < 16; ++k) {
    // batch 6 B fragments (12 global_load_b128 in flight) before consuming
#pragma unroll
    for (int h = 0; h < 2; ++h) {
      v16bf bf[6];
#pragma unroll
      for (int i = 0; i < 6; ++i)
        bf[i] = load_b_frag(wih_pk, wave * 12 + h * 6 + i, k, 16, lane);
#pragma unroll
      for (int i = 0; i < 6; ++i)
        acc[h * 6 + i] = wmma_bf16(a, bf[i], acc[h * 6 + i]);
    }
    if (k < 15) a = load_a_frag(sbf, 512, (k + 1) * 32, lo, hi);
  }
#pragma unroll
  for (int q = 0; q < 12; ++q) {
    int col = (wave * 12 + q) * 16 + lo;
#pragma unroll
    for (int r = 0; r < 8; ++r)
      gi0[(size_t)(m0 + r + 8 * hi) * 1536 + col] = acc[q][r];
  }
}

// ---- Main 32-step rollout: one WG owns 16 rows for the whole horizon ----
__global__ __launch_bounds__(256) void rollout_kernel(
    const float* __restrict__ z_dyn, const float* __restrict__ z_sem,
    const float* __restrict__ b_hh, const float* __restrict__ ln_g,
    const float* __restrict__ ln_b, const float* __restrict__ ln2_g,
    const float* __restrict__ ln2_b, const float* __restrict__ b1v,
    const float* __restrict__ b2v, const float* __restrict__ gi0,
    const float* __restrict__ tprojb, const __bf16* __restrict__ whh_pk,
    const __bf16* __restrict__ w1_pk, const __bf16* __restrict__ w2_pk,
    float* __restrict__ dyn_out, float* __restrict__ sem_out) {
  __shared__ float  sh_h[16 * 512];     // f32 h state (32 KB)
  __shared__ __bf16 sh_hbf[16 * 512];   // bf16 h (GEMM1 A)    (16 KB)
  __shared__ __bf16 sh_cat[16 * 1024];  // ln2(cat) bf16; aliased as f32 stage (32 KB)
  __shared__ __bf16 sh_hid[16 * 512];   // gelu(hid) bf16 (GEMM3 A) (16 KB)
  __shared__ float  red_a[256];
  __shared__ float  red_b[256];
  __shared__ float  mrs[64];            // [0..31] ln1 (mean,rstd) ; [32..63] ln2

  const int tid  = threadIdx.x;
  const int wave = tid >> 5, lane = tid & 31;
  const int lo = lane & 15, hi = lane >> 4;
  const int orow = tid & 15, oc0 = (tid >> 4) * 32;
  const int m0 = blockIdx.x * 16;
  const v8f z8 = {0.f, 0.f, 0.f, 0.f, 0.f, 0.f, 0.f, 0.f};

  // init h = z_dyn rows
#pragma unroll
  for (int j = 0; j < 32; ++j) {
    float h = z_dyn[(size_t)(m0 + orow) * 512 + oc0 + j];
    sh_h[orow * 512 + oc0 + j] = h;
    sh_hbf[orow * 512 + oc0 + j] = f2bf(h);
  }
  __syncthreads();

  for (int t = 0; t < HZN; ++t) {
    // ---- GEMM1: gh = h @ w_hh^T ----
    // acc index = g*4+q : gate g (0=r,1=z,2=n), col-tile q -> nt = wave*4+q+g*32
    v8f acc1[12];
#pragma unroll
    for (int q = 0; q < 12; ++q) acc1[q] = z8;
    {
      v16bf a = load_a_frag(sh_hbf, 512, 0, lo, hi);
      for (int k = 0; k < 16; ++k) {
#pragma unroll
        for (int h = 0; h < 2; ++h) {
          v16bf bf[6];
#pragma unroll
          for (int i = 0; i < 6; ++i) {
            int idx = h * 6 + i;
            int g = idx >> 2, q = idx & 3;
            bf[i] = load_b_frag(whh_pk, wave * 4 + q + g * 32, k, 16, lane);
          }
#pragma unroll
          for (int i = 0; i < 6; ++i)
            acc1[h * 6 + i] = wmma_bf16(a, bf[i], acc1[h * 6 + i]);
        }
        if (k < 15) a = load_a_frag(sh_hbf, 512, (k + 1) * 32, lo, hi);
      }
    }
    // ---- GRU gates, in WMMA accumulator layout ----
#pragma unroll
    for (int q = 0; q < 4; ++q) {
      int col = (wave * 4 + q) * 16 + lo;
      float bhr = b_hh[col], bhz = b_hh[col + 512], bhn = b_hh[col + 1024];
      float tpr = tprojb[t * 1536 + col];
      float tpz = tprojb[t * 1536 + col + 512];
      float tpn = tprojb[t * 1536 + col + 1024];
#pragma unroll
      for (int r = 0; r < 8; ++r) {
        int row = r + 8 * hi;
        const float* gp = gi0 + (size_t)(m0 + row) * 1536 + col;
        float rg  = sigm(gp[0]   + tpr + acc1[q][r]     + bhr);
        float zg  = sigm(gp[512] + tpz + acc1[4 + q][r] + bhz);
        float hn_ = acc1[8 + q][r] + bhn;
        float ng  = tanh_f(gp[1024] + tpn + rg * hn_);
        float hp  = sh_h[row * 512 + col];
        sh_h[row * 512 + col] = (1.f - zg) * ng + zg * hp;   // h_new
      }
    }
    __syncthreads();

    // ---- LN1 stats (row-ownership layout) + refresh bf16 h ----
    float hv[32];
    float s = 0.f, qq = 0.f;
#pragma unroll
    for (int j = 0; j < 32; ++j) {
      float x = sh_h[orow * 512 + oc0 + j];
      hv[j] = x; s += x; qq += x * x;
      sh_hbf[orow * 512 + oc0 + j] = f2bf(x);
    }
    red_a[(tid >> 4) * 16 + orow] = s;
    red_b[(tid >> 4) * 16 + orow] = qq;
    __syncthreads();
    if (tid < 16) {
      float ss = 0.f, sq = 0.f;
      for (int g = 0; g < 16; ++g) { ss += red_a[g * 16 + tid]; sq += red_b[g * 16 + tid]; }
      float mn = ss * (1.f / 512.f);
      float var = sq * (1.f / 512.f) - mn * mn;
      mrs[2 * tid] = mn;
      mrs[2 * tid + 1] = rsqrtf(var + 1e-5f);
    }
    __syncthreads();

    // ---- h_norm -> dyn out; LN2 stats over cat=[sem,h_norm] ----
    float m1 = mrs[2 * orow], r1 = mrs[2 * orow + 1];
    float hn2[32], sv[32];
    float s2 = 0.f, q2 = 0.f;
    {
      size_t ob = ((size_t)(m0 + orow) * HZN + t) * 512 + oc0;
#pragma unroll
      for (int j = 0; j < 32; ++j) {
        float y = (hv[j] - m1) * r1 * ln_g[oc0 + j] + ln_b[oc0 + j];
        hn2[j] = y;
        float se = z_sem[(size_t)(m0 + orow) * 512 + oc0 + j];
        sv[j] = se;
        s2 += y + se; q2 += y * y + se * se;
        dyn_out[ob + j] = y;
      }
    }
    red_a[(tid >> 4) * 16 + orow] = s2;
    red_b[(tid >> 4) * 16 + orow] = q2;
    __syncthreads();
    if (tid < 16) {
      float ss = 0.f, sq = 0.f;
      for (int g = 0; g < 16; ++g) { ss += red_a[g * 16 + tid]; sq += red_b[g * 16 + tid]; }
      float mn = ss * (1.f / 1024.f);
      float var = sq * (1.f / 1024.f) - mn * mn;
      mrs[32 + 2 * tid] = mn;
      mrs[32 + 2 * tid + 1] = rsqrtf(var + 1e-5f);
    }
    __syncthreads();

    // ---- write ln2(cat) as bf16 ----
    float m2 = mrs[32 + 2 * orow], r2 = mrs[32 + 2 * orow + 1];
#pragma unroll
    for (int j = 0; j < 32; ++j) {
      sh_cat[orow * 1024 + oc0 + j] =
          f2bf((sv[j] - m2) * r2 * ln2_g[oc0 + j] + ln2_b[oc0 + j]);
      sh_cat[orow * 1024 + 512 + oc0 + j] =
          f2bf((hn2[j] - m2) * r2 * ln2_g[512 + oc0 + j] + ln2_b[512 + oc0 + j]);
    }
    __syncthreads();

    // ---- GEMM2: hid = ln2cat @ w1^T, + b1, exact gelu ----
    v8f a2[4];
#pragma unroll
    for (int q = 0; q < 4; ++q) a2[q] = z8;
    {
      v16bf a = load_a_frag(sh_cat, 1024, 0, lo, hi);
      for (int k = 0; k < 32; ++k) {
        v16bf bf[4];
#pragma unroll
        for (int q = 0; q < 4; ++q)
          bf[q] = load_b_frag(w1_pk, wave * 4 + q, k, 32, lane);
#pragma unroll
        for (int q = 0; q < 4; ++q) a2[q] = wmma_bf16(a, bf[q], a2[q]);
        if (k < 31) a = load_a_frag(sh_cat, 1024, (k + 1) * 32, lo, hi);
      }
    }
#pragma unroll
    for (int q = 0; q < 4; ++q) {
      int col = (wave * 4 + q) * 16 + lo;
      float bb = b1v[col];
#pragma unroll
      for (int r = 0; r < 8; ++r) {
        float x = a2[q][r] + bb;
        float gl = 0.5f * x * (1.f + erff(x * 0.70710678118654752f));
        sh_hid[(r + 8 * hi) * 512 + col] = f2bf(gl);
      }
    }
    __syncthreads();

    // ---- GEMM3: hid @ w2^T -> stage (aliases sh_cat, safe after barrier) ----
    v8f a3[4];
#pragma unroll
    for (int q = 0; q < 4; ++q) a3[q] = z8;
    {
      v16bf a = load_a_frag(sh_hid, 512, 0, lo, hi);
      for (int k = 0; k < 16; ++k) {
        v16bf bf[4];
#pragma unroll
        for (int q = 0; q < 4; ++q)
          bf[q] = load_b_frag(w2_pk, wave * 4 + q, k, 16, lane);
#pragma unroll
        for (int q = 0; q < 4; ++q) a3[q] = wmma_bf16(a, bf[q], a3[q]);
        if (k < 15) a = load_a_frag(sh_hid, 512, (k + 1) * 32, lo, hi);
      }
    }
    float* stage = (float*)sh_cat;
#pragma unroll
    for (int q = 0; q < 4; ++q) {
      int col = (wave * 4 + q) * 16 + lo;
#pragma unroll
      for (int r = 0; r < 8; ++r)
        stage[(r + 8 * hi) * 512 + col] = a3[q][r];
    }
    __syncthreads();

    // ---- sems out = sem + hid@w2^T + b2 (coalesced) ----
    {
      size_t ob = ((size_t)(m0 + orow) * HZN + t) * 512 + oc0;
#pragma unroll
      for (int j = 0; j < 32; ++j)
        sem_out[ob + j] = stage[orow * 512 + oc0 + j] + sv[j] + b2v[oc0 + j];
    }
    __syncthreads();
  }
}

extern "C" void kernel_launch(void* const* d_in, const int* in_sizes, int n_in,
                              void* d_out, int out_size, void* d_ws, size_t ws_size,
                              hipStream_t stream) {
  const float* z_dyn = (const float*)d_in[0];
  const float* z_sem = (const float*)d_in[1];
  const float* te_w  = (const float*)d_in[2];
  const float* w_ih  = (const float*)d_in[3];
  const float* w_hh  = (const float*)d_in[4];
  const float* b_ih  = (const float*)d_in[5];
  const float* b_hh  = (const float*)d_in[6];
  const float* ln_g  = (const float*)d_in[7];
  const float* ln_b  = (const float*)d_in[8];
  const float* ln2_g = (const float*)d_in[9];
  const float* ln2_b = (const float*)d_in[10];
  const float* w1    = (const float*)d_in[11];
  const float* b1v   = (const float*)d_in[12];
  const float* w2    = (const float*)d_in[13];
  const float* b2v   = (const float*)d_in[14];

  char* ws = (char*)d_ws;
  float*  gi0    = (float*)(ws);                 // 2048*1536 f32 = 12,582,912 B
  float*  tprojb = (float*)(ws + 12582912);      // 32*1536 f32   =    196,608 B
  __bf16* whh_pk = (__bf16*)(ws + 12779520);     // 1536*512 bf16 =  1,572,864 B
  __bf16* wih_pk = (__bf16*)(ws + 14352384);     // 1536*512 bf16 =  1,572,864 B
  __bf16* w1_pk  = (__bf16*)(ws + 15925248);     // 512*1024 bf16 =  1,048,576 B
  __bf16* w2_pk  = (__bf16*)(ws + 16973824);     // 512*512  bf16 =    524,288 B

  pack_kernel<<<768, 256, 0, stream>>>(w_hh, whh_pk, 1536, 512);
  pack_kernel<<<768, 256, 0, stream>>>(w_ih, wih_pk, 1536, 512);
  pack_kernel<<<512, 256, 0, stream>>>(w1, w1_pk, 512, 1024);
  pack_kernel<<<256, 256, 0, stream>>>(w2, w2_pk, 512, 512);
  tproj_kernel<<<192, 256, 0, stream>>>(te_w, w_ih, b_ih, tprojb);
  gi0_kernel<<<128, 256, 0, stream>>>(z_sem, wih_pk, gi0);

  float* dyn_out = (float*)d_out;
  float* sem_out = dyn_out + (size_t)2048 * HZN * 512;
  rollout_kernel<<<128, 256, 0, stream>>>(z_dyn, z_sem, b_hh, ln_g, ln_b, ln2_g,
                                          ln2_b, b1v, b2v, gi0, tprojb, whh_pk,
                                          w1_pk, w2_pk, dyn_out, sem_out);
}
